// VQPatternMatrix_80616536145991
// MI455X (gfx1250) — compile-verified, matrix operand order checked by reading, not run on hardware
//
#include <hip/hip_runtime.h>
#include <hip/hip_bf16.h>

#define D_MODEL 256
#define N_PAT   32

typedef __attribute__((ext_vector_type(16))) __bf16 v16bf;
typedef __attribute__((ext_vector_type(2)))  __bf16 bf16x2;
typedef __attribute__((ext_vector_type(8)))  float  v8f;
typedef __attribute__((ext_vector_type(2)))  float  v2f;

// ---------------------------------------------------------------------------
// Prep 1: W_q -> bf16 in ws; new_ema = 0.99*ema + 0.01*patterns; zero histogram
// ---------------------------------------------------------------------------
__global__ __launch_bounds__(256) void vq_prep1(const float* __restrict__ W,
                                                const float* __restrict__ patterns,
                                                const float* __restrict__ ema,
                                                unsigned short* __restrict__ Wb,
                                                float* __restrict__ ema_out,
                                                unsigned* __restrict__ hist) {
  int i = blockIdx.x * 256 + threadIdx.x;               // 0 .. 65535
  __bf16 b = (__bf16)W[i];
  Wb[i] = __builtin_bit_cast(unsigned short, b);
  if (i < N_PAT * D_MODEL)
    ema_out[i] = 0.99f * ema[i] + 0.01f * patterns[i];
  if (i < N_PAT) hist[i] = 0u;
}

// ---------------------------------------------------------------------------
// Prep 2: k = l2norm(patterns) rows -> bf16 (one wave per pattern row)
// ---------------------------------------------------------------------------
__global__ __launch_bounds__(32) void vq_prep2(const float* __restrict__ patterns,
                                               unsigned short* __restrict__ Pnb) {
  int p = blockIdx.x, lane = threadIdx.x;
  float s = 0.f;
  float v[8];
#pragma unroll
  for (int e = 0; e < 8; ++e) { v[e] = patterns[p * D_MODEL + lane * 8 + e]; s += v[e] * v[e]; }
#pragma unroll
  for (int m = 1; m < 32; m <<= 1) s += __shfl_xor(s, m, 32);
  float inv = 1.0f / fmaxf(sqrtf(s), 1e-12f);
#pragma unroll
  for (int e = 0; e < 8; ++e) {
    __bf16 b = (__bf16)(v[e] * inv);
    Pnb[p * D_MODEL + lane * 8 + e] = __builtin_bit_cast(unsigned short, b);
  }
}

// ---------------------------------------------------------------------------
// Main fused kernel: 2 waves / block, 16 tokens / wave (round-1 structure,
// which produced clean b128-clause + WMMA pipelining with <256 VGPRs).
// LayerNorm phase vectorized to float4 LDS traffic.
// ---------------------------------------------------------------------------
__global__ __launch_bounds__(64) void vq_main(const float* __restrict__ x,
                                              const float* __restrict__ b_q,
                                              const float* __restrict__ ln_w,
                                              const float* __restrict__ ln_b,
                                              const float* __restrict__ patterns,
                                              const unsigned short* __restrict__ Wb,
                                              const unsigned short* __restrict__ Pnb,
                                              unsigned* __restrict__ hist_g,
                                              const int* __restrict__ epoch_p,
                                              const int* __restrict__ total_p,
                                              float* __restrict__ emb_out,
                                              float* __restrict__ soft_out,
                                              float* __restrict__ logits_out,
                                              float* __restrict__ hard_out) {
  __shared__ float    h_lds[2][16][260];   // 1040 B rows -> 16B aligned
  __shared__ float    lg_lds[2][16][36];
  __shared__ float    inv_lds[2][16];
  __shared__ unsigned hist_lds[N_PAT];

  const int tid  = threadIdx.x;
  const int lane = tid & 31;
  const int wid  = tid >> 5;
  if (tid < N_PAT) hist_lds[tid] = 0u;
  __syncthreads();

  const int t0    = (blockIdx.x * 2 + wid) * 16;
  const int rA    = lane & 15;            // frag row / column
  const int lh    = lane >> 4;            // lane half
  const int khalf = lh * 8;               // 16-bit frag K offset per lane half

  // temperature schedule from device scalars
  int ep = *epoch_p, te = *total_p;
  float prog = (te > 0) ? ((float)ep / (float)te) : 0.f;
  float inv_temp = 1.0f / fmaxf(0.5f, 1.0f - 0.5f * prog);

  // ---- Phase A: h = x @ W^T + b via bf16 WMMA (f32 accum) ------------------
  v16bf a[8];
  {
    const float2* x2 = (const float2*)(x + (size_t)(t0 + rA) * D_MODEL);
#pragma unroll
    for (int c = 0; c < 8; ++c) {
#pragma unroll
      for (int v = 0; v < 8; ++v) {
        int kp = 32 * c + (v & 3) * 2 + (v >> 2) * 16 + khalf;
        float2 f = x2[kp >> 1];
        a[c][2 * v]     = (__bf16)f.x;
        a[c][2 * v + 1] = (__bf16)f.y;
      }
    }
  }
  {
    const unsigned* Wb32 = (const unsigned*)Wb;
    for (int j = 0; j < 16; ++j) {                 // 16 column tiles of h
      float bv = b_q[j * 16 + rA];
      v8f acc;
#pragma unroll
      for (int r = 0; r < 8; ++r) acc[r] = bv;
#pragma unroll
      for (int c = 0; c < 8; ++c) {
        v16bf bfrag;
#pragma unroll
        for (int v = 0; v < 8; ++v) {
          int kp = 32 * c + (v & 3) * 2 + (v >> 2) * 16 + khalf;
          unsigned packed = Wb32[((j * 16 + rA) * D_MODEL + kp) >> 1];
          bf16x2 pr = __builtin_bit_cast(bf16x2, packed);
          bfrag[2 * v]     = pr.x;
          bfrag[2 * v + 1] = pr.y;
        }
        acc = __builtin_amdgcn_wmma_f32_16x16x32_bf16(false, a[c], false, bfrag,
                                                      (short)0, acc, false, false);
      }
#pragma unroll
      for (int r = 0; r < 8; ++r)
        h_lds[wid][r + (lh << 3)][j * 16 + rA] = acc[r];
    }
  }
  __syncthreads();

  // ---- Phase B: LayerNorm + L2 norm (2 lanes per token, float4 LDS) --------
  {
    int t = lane >> 1, half = lane & 1, f0 = half * 128;
    const float4* hr   = (const float4*)&h_lds[wid][t][f0];
    float4*       hw   = (float4*)&h_lds[wid][t][f0];
    const float4* lnw4 = (const float4*)(ln_w + f0);
    const float4* lnb4 = (const float4*)(ln_b + f0);
    float s = 0.f, ss = 0.f;
    for (int i = 0; i < 32; ++i) {
      float4 h = hr[i];
      s  += h.x + h.y + h.z + h.w;
      ss += h.x * h.x + h.y * h.y + h.z * h.z + h.w * h.w;
    }
    s  += __shfl_xor(s, 1, 32);
    ss += __shfl_xor(ss, 1, 32);
    float mu   = s * (1.0f / 256.0f);
    float var  = ss * (1.0f / 256.0f) - mu * mu;
    float rstd = rsqrtf(var + 1e-5f);
    float g2 = 0.f;
    for (int i = 0; i < 32; ++i) {
      float4 h = hr[i], w = lnw4[i], bb = lnb4[i], g;
      g.x = (h.x - mu) * rstd * w.x + bb.x;
      g.y = (h.y - mu) * rstd * w.y + bb.y;
      g.z = (h.z - mu) * rstd * w.z + bb.z;
      g.w = (h.w - mu) * rstd * w.w + bb.w;
      hw[i] = g;
      g2 += g.x * g.x + g.y * g.y + g.z * g.z + g.w * g.w;
    }
    g2 += __shfl_xor(g2, 1, 32);
    float inv = 1.0f / fmaxf(sqrtf(g2), 1e-12f);
    if (half == 0) inv_lds[wid][t] = inv;
  }
  __syncthreads();

  // ---- Phase C: logits = (q . k) / temp via bf16 WMMA ----------------------
  {
    v16bf qa[8];
    float invq = inv_lds[wid][rA];
    const float2* qrow = (const float2*)&h_lds[wid][rA][0];
#pragma unroll
    for (int c = 0; c < 8; ++c) {
#pragma unroll
      for (int v = 0; v < 8; ++v) {
        int kp = 32 * c + (v & 3) * 2 + (v >> 2) * 16 + khalf;
        float2 q2 = qrow[kp >> 1];
        qa[c][2 * v]     = (__bf16)(q2.x * invq);
        qa[c][2 * v + 1] = (__bf16)(q2.y * invq);
      }
    }
    const unsigned* Pn32 = (const unsigned*)Pnb;
#pragma unroll
    for (int jn = 0; jn < 2; ++jn) {
      v8f acc;
#pragma unroll
      for (int r = 0; r < 8; ++r) acc[r] = 0.f;
#pragma unroll
      for (int c = 0; c < 8; ++c) {
        v16bf bfrag;
#pragma unroll
        for (int v = 0; v < 8; ++v) {
          int kp = 32 * c + (v & 3) * 2 + (v >> 2) * 16 + khalf;
          unsigned packed = Pn32[((jn * 16 + rA) * D_MODEL + kp) >> 1];
          bf16x2 pr = __builtin_bit_cast(bf16x2, packed);
          bfrag[2 * v]     = pr.x;
          bfrag[2 * v + 1] = pr.y;
        }
        acc = __builtin_amdgcn_wmma_f32_16x16x32_bf16(false, qa[c], false, bfrag,
                                                      (short)0, acc, false, false);
      }
#pragma unroll
      for (int r = 0; r < 8; ++r) {
        int M = r + (lh << 3);
        float lg = acc[r] * inv_temp;
        lg_lds[wid][M][jn * 16 + rA] = lg;
        logits_out[(size_t)(t0 + M) * N_PAT + jn * 16 + rA] = lg;
      }
    }
  }
  __syncthreads();

  // ---- Softmax + argmax + histogram (2 lanes per token) --------------------
  {
    int t = lane >> 1, half = lane & 1, p0 = half * 16;
    float m = -3.0e38f; int bi = p0;
    for (int i = 0; i < 16; ++i) {
      float v = lg_lds[wid][t][p0 + i];
      if (v > m) { m = v; bi = p0 + i; }
    }
    float om = __shfl_xor(m, 1, 32);
    int  obi = __shfl_xor(bi, 1, 32);
    float mm = fmaxf(m, om);
    int bidx = (om > m || (om == m && obi < bi)) ? obi : bi;
    float ev[16], es = 0.f;
    for (int i = 0; i < 16; ++i) {
      ev[i] = expf(lg_lds[wid][t][p0 + i] - mm);
      es += ev[i];
    }
    es += __shfl_xor(es, 1, 32);
    float rs = 1.0f / es;
    for (int i = 0; i < 16; ++i) {
      float pr = ev[i] * rs;
      lg_lds[wid][t][p0 + i] = pr;
      soft_out[(size_t)(t0 + t) * N_PAT + p0 + i] = pr;
    }
    if (half == 0) {
      hard_out[t0 + t] = (float)bidx;
      atomicAdd(&hist_lds[bidx], 1u);
    }
  }
  __syncthreads();

  // ---- Phase D: emb = soft @ patterns via f32 WMMA (16x16x4) ---------------
  {
    v2f sa[8];
#pragma unroll
    for (int c4 = 0; c4 < 8; ++c4) {
      int kb = 4 * c4 + lh * 2;
      sa[c4].x = lg_lds[wid][rA][kb];
      sa[c4].y = lg_lds[wid][rA][kb + 1];
    }
    for (int jn = 0; jn < 16; ++jn) {
      v8f acc;
#pragma unroll
      for (int r = 0; r < 8; ++r) acc[r] = 0.f;
#pragma unroll
      for (int c4 = 0; c4 < 8; ++c4) {
        int kb = 4 * c4 + lh * 2;
        v2f bfr;
        bfr.x = patterns[(size_t)kb * D_MODEL + jn * 16 + rA];
        bfr.y = patterns[(size_t)(kb + 1) * D_MODEL + jn * 16 + rA];
        acc = __builtin_amdgcn_wmma_f32_16x16x4_f32(false, sa[c4], false, bfr,
                                                    (short)0, acc, false, false);
      }
#pragma unroll
      for (int r = 0; r < 8; ++r)
        emb_out[(size_t)(t0 + r + (lh << 3)) * D_MODEL + jn * 16 + rA] = acc[r];
    }
  }

  // flush per-block histogram (ordered by the pre-Phase-D barrier)
  if (tid < N_PAT && hist_lds[tid]) atomicAdd(&hist_g[tid], hist_lds[tid]);
}

// ---------------------------------------------------------------------------
// Finalize: new_usage = 0.9*old + 0.1 * hist/(sum+1e-8)
// ---------------------------------------------------------------------------
__global__ __launch_bounds__(32) void vq_finalize(const unsigned* __restrict__ hist,
                                                  const float* __restrict__ usage_old,
                                                  float* __restrict__ usage_out) {
  int lane = threadIdx.x;
  float c = (float)hist[lane];
  float tot = c;
#pragma unroll
  for (int m = 1; m < 32; m <<= 1) tot += __shfl_xor(tot, m, 32);
  float u = c / (tot + 1e-8f);
  usage_out[lane] = 0.9f * usage_old[lane] + 0.1f * u;
}

extern "C" void kernel_launch(void* const* d_in, const int* in_sizes, int n_in,
                              void* d_out, int out_size, void* d_ws, size_t ws_size,
                              hipStream_t stream) {
  const float* x        = (const float*)d_in[0];
  const float* W_q      = (const float*)d_in[1];
  const float* b_q      = (const float*)d_in[2];
  const float* ln_w     = (const float*)d_in[3];
  const float* ln_b     = (const float*)d_in[4];
  const float* patterns = (const float*)d_in[5];
  const float* usage_in = (const float*)d_in[6];
  const float* ema_in   = (const float*)d_in[7];
  const int*   epoch_p  = (const int*)d_in[8];
  const int*   total_p  = (const int*)d_in[9];

  const size_t nT = (size_t)in_sizes[0] / D_MODEL;   // 131072 tokens

  float* out        = (float*)d_out;
  float* emb_out    = out;
  float* soft_out   = emb_out + nT * D_MODEL;
  float* logits_out = soft_out + nT * N_PAT;
  float* hard_out   = logits_out + nT * N_PAT;
  float* usage_out  = hard_out + nT;
  float* ema_out    = usage_out + N_PAT;

  char* ws = (char*)d_ws;
  unsigned short* Wb   = (unsigned short*)ws;                       // 128 KB
  unsigned short* Pnb  = (unsigned short*)(ws + 131072);            //  16 KB
  unsigned*       hist = (unsigned*)(ws + 131072 + 16384);          // 128 B

  vq_prep1<<<(D_MODEL * D_MODEL) / 256, 256, 0, stream>>>(W_q, patterns, ema_in,
                                                          Wb, ema_out, hist);
  vq_prep2<<<N_PAT, 32, 0, stream>>>(patterns, Pnb);

  vq_main<<<(unsigned)(nT / 32), 64, 0, stream>>>(x, b_q, ln_w, ln_b, patterns,
                                                  Wb, Pnb, hist, epoch_p, total_p,
                                                  emb_out, soft_out, logits_out,
                                                  hard_out);

  vq_finalize<<<1, 32, 0, stream>>>(hist, usage_in, usage_out);
}
